// CTMCritic_6597069766976
// MI455X (gfx1250) — matrix operand order repaired and checked by reading, not compiled, not run on hardware
//
#include <hip/hip_runtime.h>
#include <hip/hip_bf16.h>
#include <math.h>

typedef _Float16 f16;
typedef __attribute__((ext_vector_type(16))) _Float16 v16h;
typedef __attribute__((ext_vector_type(8)))  _Float16 v8h;
typedef __attribute__((ext_vector_type(8)))  float    v8f;

#define BQ    256      // batch
#define TICKS 16
#define DM    1024     // D_MODEL
#define DIN   512      // D_IN
#define SRA_  32896    // 256*257/2  (== SRO)
#define SSEQ  72       // 64 traj tokens + 8 agents
#define DSYN  1536     // D_IN + D_MODEL
#define DSYN2 2048     // 2*D_MODEL

// Async global->LDS 16B copy (gfx1250 VGLOBAL, tracked by ASYNCcnt).
// vdst = LDS byte offset (addr[31:0] of the generic pointer, ISA 10.2),
// vaddr = 64-bit global address.
__device__ __forceinline__ void async_copy_b128(unsigned lds_off, const void* gptr) {
    asm volatile("global_load_async_to_lds_b128 %0, %1, off"
                 :: "v"(lds_off), "v"((unsigned long long)(uintptr_t)gptr)
                 : "memory");
}
__device__ __forceinline__ void wait_async_le2() {
    asm volatile("s_wait_asynccnt 0x2" ::: "memory");
}
__device__ __forceinline__ void wait_async_0() {
    asm volatile("s_wait_asynccnt 0x0" ::: "memory");
}

// ---------------------------------------------------------------------------
// WMMA GEMM: out = A(MxK,f16) * W(KxN given as WT = NxK,f16) [+ bias], f32 acc.
// 256 threads = 8 waves; block tile 64(M)x64(N); K-step 32; double-buffered
// LDS tiles filled with GLOBAL_LOAD_ASYNC_TO_LDS_B128.
// kSplit>1: interleaved split-K, f32 atomic accumulate into outF (pre-zeroed).
// ---------------------------------------------------------------------------
__global__ void __launch_bounds__(256)
k_gemm_wmma(const f16* __restrict__ A, const f16* __restrict__ BT,
            const float* __restrict__ bias,
            float* __restrict__ outF, f16* __restrict__ outH,
            int M, int N, int K, int ldOut, int kSplit)
{
    __shared__ f16 As[2][64][32];   // [buf][m][k]
    __shared__ f16 Bs[2][64][32];   // [buf][n][k]

    const int tid    = threadIdx.x;
    const int mBlock = blockIdx.y * 64;
    const int nBlock = blockIdx.x * 64;
    const int z      = blockIdx.z;
    const int lane   = tid & 31;
    const int wid    = tid >> 5;
    const int mBase  = (wid >> 2) * 32;
    const int nBase  = (wid & 3) * 16;
    const int lh     = lane & 15;
    const int hi     = lane >> 4;

    const int row = tid >> 2;        // 0..63
    const int col = (tid & 3) * 8;   // 0,8,16,24  (16B granules)

    const int kStep = kSplit * 32;
    const f16* aSrc = A  + (size_t)(mBlock + row) * K + z * 32 + col;
    const f16* bSrc = BT + (size_t)(nBlock + row) * K + z * 32 + col;

    const unsigned ldsA0 = (unsigned)(uintptr_t)&As[0][row][col];
    const unsigned ldsA1 = (unsigned)(uintptr_t)&As[1][row][col];
    const unsigned ldsB0 = (unsigned)(uintptr_t)&Bs[0][row][col];
    const unsigned ldsB1 = (unsigned)(uintptr_t)&Bs[1][row][col];

    v8f acc0 = {}; v8f acc1 = {};

    const int nSteps = (K - z * 32 + kStep - 1) / kStep;

    // prologue: tile 0 in flight
    async_copy_b128(ldsA0, aSrc);
    async_copy_b128(ldsB0, bSrc);

    for (int i = 0; i < nSteps; ++i) {
        const int buf = i & 1;
        if (i + 1 < nSteps) {
            const f16* aN = aSrc + (size_t)(i + 1) * kStep;
            const f16* bN = bSrc + (size_t)(i + 1) * kStep;
            async_copy_b128(buf ? ldsA0 : ldsA1, aN);
            async_copy_b128(buf ? ldsB0 : ldsB1, bN);
            wait_async_le2();     // tile i landed; tile i+1 still streaming
        } else {
            wait_async_0();
        }
        __syncthreads();

        // A fragment: lane = m + 16*hi; elems 0..7 = K[8hi..8hi+7], 8..15 = K[16+8hi..23+8hi]
        const f16* ar0 = &As[buf][mBase + lh][0];
        const f16* ar1 = &As[buf][mBase + 16 + lh][0];
        v8h a0lo = *reinterpret_cast<const v8h*>(ar0 + 8 * hi);
        v8h a0hi = *reinterpret_cast<const v8h*>(ar0 + 16 + 8 * hi);
        v8h a1lo = *reinterpret_cast<const v8h*>(ar1 + 8 * hi);
        v8h a1hi = *reinterpret_cast<const v8h*>(ar1 + 16 + 8 * hi);
        // B fragment: lane = n + 16*hi; elems 0..7 = K[8hi..8hi+7], 8..15 = K[16+8hi..23+8hi]
        const f16* br = &Bs[buf][nBase + lh][0];
        v8h blo = *reinterpret_cast<const v8h*>(br + 8 * hi);
        v8h bhi = *reinterpret_cast<const v8h*>(br + 16 + 8 * hi);

        v16h af0, af1, bf;
        #pragma unroll
        for (int q = 0; q < 8; ++q) {
            af0[q] = a0lo[q]; af0[q + 8] = a0hi[q];
            af1[q] = a1lo[q]; af1[q + 8] = a1hi[q];
            bf[q]  = blo[q];  bf[q + 8]  = bhi[q];
        }
        acc0 = __builtin_amdgcn_wmma_f32_16x16x32_f16(false, af0, false, bf, (short)0, acc0, false, false);
        acc1 = __builtin_amdgcn_wmma_f32_16x16x32_f16(false, af1, false, bf, (short)0, acc1, false, false);
        __syncthreads();   // safe to overwrite buf at iteration i+1's issue
    }

    const int nG = nBlock + nBase + lh;
    if (kSplit > 1) {
        #pragma unroll
        for (int r = 0; r < 8; ++r) {
            int m0 = mBlock + mBase + r + 8 * hi;
            atomicAdd(&outF[(size_t)m0 * ldOut + nG], acc0[r]);
            atomicAdd(&outF[(size_t)(m0 + 16) * ldOut + nG], acc1[r]);
        }
    } else {
        float bv = bias ? bias[nG] : 0.f;
        #pragma unroll
        for (int r = 0; r < 8; ++r) {
            int m0 = mBlock + mBase + r + 8 * hi;
            int m1 = m0 + 16;
            float v0 = acc0[r] + bv, v1 = acc1[r] + bv;
            if (outF) { outF[(size_t)m0 * ldOut + nG] = v0; outF[(size_t)m1 * ldOut + nG] = v1; }
            if (outH) { outH[(size_t)m0 * ldOut + nG] = (f16)v0; outH[(size_t)m1 * ldOut + nG] = (f16)v1; }
        }
    }
}

// --------------------------- setup kernels ---------------------------------
__global__ void k_tables(int* i0, int* i1) {
    int i = threadIdx.x;                      // 256 threads, 1 block
    int off = i * 256 - (i * (i - 1)) / 2;    // triu row offset
    for (int t = 0; t < 256 - i; ++t) { i0[off + t] = i; i1[off + t] = i + t; }
}

__global__ void k_decay_init(const float* dpa, const float* dpo,
                             float* ra, float* ro, float* dba, float* dbo) {
    int s = blockIdx.x * 256 + threadIdx.x;
    if (s < SRA_) {
        ra[s] = __expf(-fminf(fmaxf(dpa[s], 0.f), 15.f));
        ro[s] = __expf(-fminf(fmaxf(dpo[s], 0.f), 15.f));
        dba[s] = 0.f;        // db_a0 = 0
        dbo[s] = 1.f;        // db_o0 = 1
    }
}

__global__ void k_init_state(const float* sas, const float* st,
                             float* act, float* trace, float* q_all, float* c_all) {
    size_t idx = (size_t)blockIdx.x * 256 + threadIdx.x;
    if (idx < (size_t)BQ * DM * 32) trace[idx] = st[idx % (DM * 32)];
    if (idx < (size_t)BQ * DM)      act[idx]   = sas[idx % DM];
    if (idx < (size_t)BQ * TICKS)   { q_all[idx] = 0.f; c_all[idx] = 0.f; }
}

__global__ void k_init_da(const float* sas, const int* i0, const int* i1,
                          float* da_a, float* da_o) {
    size_t idx = (size_t)blockIdx.x * 256 + threadIdx.x;
    if (idx >= (size_t)BQ * SRA_) return;
    int s = (int)(idx % SRA_);
    da_a[idx] = 0.f;
    da_o[idx] = sas[i0[s]] * sas[i1[s]];   // da_o0 = pw(act0[:, :NSO])
}

// convert f32 KxN weight -> f16 NxK (transposed, so GEMM B staging is contiguous)
__global__ void __launch_bounds__(256)
k_cvt_f16T(const float* __restrict__ src, f16* __restrict__ dst, int K, int N) {
    __shared__ float tile[32][33];
    int n0 = blockIdx.x * 32, k0 = blockIdx.y * 32;
    int tx = threadIdx.x, ty = threadIdx.y;          // (32,8)
    for (int i = ty; i < 32; i += 8)
        tile[i][tx] = src[(size_t)(k0 + i) * N + n0 + tx];
    __syncthreads();
    for (int i = ty; i < 32; i += 8)
        dst[(size_t)(n0 + i) * K + k0 + tx] = (f16)tile[tx][i];
}

__global__ void k_zero_f32(float* p, size_t n) {
    size_t i = (size_t)blockIdx.x * 256 + threadIdx.x;
    if (i < n) p[i] = 0.f;
}

// --------------------------- kv encoder ------------------------------------
__global__ void __launch_bounds__(128)
k_build_kv(const float* __restrict__ traj, const float* __restrict__ obs,
           const float* __restrict__ acts,
           const float* __restrict__ Wt, const float* __restrict__ bt,
           const float* __restrict__ Wa, const float* __restrict__ ba,
           const float* __restrict__ g, const float* __restrict__ bb,
           f16* __restrict__ kv)
{
    __shared__ float feat[72];
    __shared__ float row[512];
    __shared__ float red[128];
    int b = blockIdx.x / SSEQ, s = blockIdx.x % SSEQ, tid = threadIdx.x;
    const float* W; const float* bias; int Klen;
    if (s < 64) {
        if (tid < 32) {
            int a = tid >> 2, c = tid & 3;
            const float* p = traj + (((size_t)b * 64 + s) * 8 + a) * 2;
            float v;
            if (c < 2) v = p[c];
            else {
                int d = c - 2;
                v = (s == 0) ? 0.f
                    : (p[d] - traj[(((size_t)b * 64 + (s - 1)) * 8 + a) * 2 + d]) * (1.0f / 0.8f);
            }
            feat[tid] = v;
        }
        W = Wt; bias = bt; Klen = 32;
    } else {
        int a = s - 64;
        if (tid < 64)      feat[tid] = obs[((size_t)b * 8 + a) * 64 + tid];
        else if (tid < 72) feat[tid] = acts[((size_t)b * 8 + a) * 8 + (tid - 64)];
        W = Wa; bias = ba; Klen = 72;
    }
    __syncthreads();
    for (int d = tid; d < 512; d += 128) {
        float acc = bias[d];
        for (int k = 0; k < Klen; ++k) acc += feat[k] * W[k * 512 + d];
        row[d] = acc;
    }
    __syncthreads();
    float p = 0;
    for (int d = tid; d < 512; d += 128) p += row[d];
    red[tid] = p; __syncthreads();
    for (int st = 64; st > 0; st >>= 1) { if (tid < st) red[tid] += red[tid + st]; __syncthreads(); }
    float mean = red[0] * (1.f / 512.f);
    __syncthreads();
    p = 0;
    for (int d = tid; d < 512; d += 128) { float dd = row[d] - mean; p += dd * dd; }
    red[tid] = p; __syncthreads();
    for (int st = 64; st > 0; st >>= 1) { if (tid < st) red[tid] += red[tid + st]; __syncthreads(); }
    float rstd = rsqrtf(red[0] * (1.f / 512.f) + 1e-6f);
    for (int d = tid; d < 512; d += 128)
        kv[((size_t)b * SSEQ + s) * 512 + d] = (f16)((row[d] - mean) * rstd * g[d] + bb[d]);
}

// --------------------------- per-tick kernels ------------------------------
__global__ void k_db(const float* ra, const float* ro, float* dba, float* dbo) {
    int s = blockIdx.x * 256 + threadIdx.x;
    if (s < SRA_) { dba[s] = ra[s] * dba[s] + 1.f; dbo[s] = ro[s] * dbo[s] + 1.f; }
}

__global__ void __launch_bounds__(256)
k_sync_a(const float* __restrict__ act, const int* __restrict__ i0, const int* __restrict__ i1,
         const float* __restrict__ r, const float* __restrict__ db,
         float* __restrict__ da, f16* __restrict__ synch)
{
    int s = blockIdx.x * 256 + threadIdx.x;
    int b = blockIdx.y;
    if (s >= SRA_) return;
    const float* ab = act + (size_t)b * DM + (DM - 256);     // act[:, -NSA:]
    float v = ab[i0[s]] * ab[i1[s]];
    size_t idx = (size_t)b * SRA_ + s;
    float d = r[s] * da[idx] + v;
    da[idx] = d;
    synch[idx] = (f16)(d * rsqrtf(db[s]));
}

__global__ void k_bias_cvt(const float* acc, const float* bias, f16* out, int N, size_t n) {
    size_t i = (size_t)blockIdx.x * 256 + threadIdx.x;
    if (i < n) out[i] = (f16)(acc[i] + bias[i % N]);
}

__global__ void __launch_bounds__(256)
k_attention(const float* __restrict__ qh, const f16* __restrict__ kh,
            const f16* __restrict__ vh, f16* __restrict__ ao)
{
    __shared__ float sq[512];
    __shared__ float lg[8][72];
    __shared__ float mh[8], ih[8];
    int b = blockIdx.x, tid = threadIdx.x;
    int h = tid >> 5, lane = tid & 31;
    for (int d = tid; d < 512; d += 256) sq[d] = qh[(size_t)b * 512 + d];
    __syncthreads();
    for (int s = lane; s < SSEQ; s += 32) {
        const f16* kp = kh + ((size_t)b * SSEQ + s) * 512 + h * 64;
        float acc = 0;
        for (int d = 0; d < 64; ++d) acc += sq[h * 64 + d] * (float)kp[d];
        lg[h][s] = acc * 0.125f;                 // 1/sqrt(HD=64)
    }
    __syncthreads();
    if (lane == 0) {
        float m = -1e30f;
        for (int s = 0; s < SSEQ; ++s) m = fmaxf(m, lg[h][s]);
        mh[h] = m;
    }
    __syncthreads();
    for (int s = lane; s < SSEQ; s += 32) lg[h][s] = __expf(lg[h][s] - mh[h]);
    __syncthreads();
    if (lane == 0) {
        float sm = 0;
        for (int s = 0; s < SSEQ; ++s) sm += lg[h][s];
        ih[h] = 1.0f / sm;
    }
    __syncthreads();
    #pragma unroll
    for (int j = 0; j < 2; ++j) {
        int d = lane * 2 + j;
        float acc = 0;
        for (int s = 0; s < SSEQ; ++s)
            acc += lg[h][s] * (float)vh[((size_t)b * SSEQ + s) * 512 + h * 64 + d];
        ao[(size_t)b * 512 + h * 64 + d] = (f16)(acc * ih[h]);
    }
}

__global__ void k_fill_act(const float* act, f16* concat) {
    size_t i = (size_t)blockIdx.x * 256 + threadIdx.x;
    if (i < (size_t)BQ * DM) {
        int b = (int)(i >> 10), n = (int)(i & 1023);
        concat[(size_t)b * DSYN + DIN + n] = (f16)act[i];
    }
}

__global__ void __launch_bounds__(256)
k_glu_ln_trace(const float* __restrict__ pre, const float* __restrict__ g,
               const float* __restrict__ bb, float* __restrict__ trace, int slot)
{
    __shared__ float x[DM];
    __shared__ float red[256];
    int b = blockIdx.x, tid = threadIdx.x;
    for (int n = tid; n < DM; n += 256) {
        float a  = pre[(size_t)b * DSYN2 + n];
        float gt = pre[(size_t)b * DSYN2 + DM + n];
        x[n] = a * (1.f / (1.f + __expf(-gt)));
    }
    __syncthreads();
    float p = 0;
    for (int n = tid; n < DM; n += 256) p += x[n];
    red[tid] = p; __syncthreads();
    for (int st = 128; st > 0; st >>= 1) { if (tid < st) red[tid] += red[tid + st]; __syncthreads(); }
    float mean = red[0] * (1.f / (float)DM);
    __syncthreads();
    p = 0;
    for (int n = tid; n < DM; n += 256) { float d = x[n] - mean; p += d * d; }
    red[tid] = p; __syncthreads();
    for (int st = 128; st > 0; st >>= 1) { if (tid < st) red[tid] += red[tid + st]; __syncthreads(); }
    float rstd = rsqrtf(red[0] * (1.f / (float)DM) + 1e-6f);
    for (int n = tid; n < DM; n += 256)
        trace[((size_t)b * DM + n) * 32 + slot] = (x[n] - mean) * rstd * g[n] + bb[n];
}

__global__ void __launch_bounds__(256)
k_trace_mlp(const float* __restrict__ trace, const float* __restrict__ w1,
            const float* __restrict__ b1, const float* __restrict__ w2,
            const float* __restrict__ b2, float* __restrict__ act, int tick)
{
    __shared__ float sw1[1024];   // [m][o]
    __shared__ float sb1[32];
    __shared__ float sw2[32];     // [h][o]
    __shared__ float sb2[2];
    int n = blockIdx.x, b = threadIdx.x;
    for (int i = b; i < 1024; i += 256) sw1[i] = w1[(size_t)n * 1024 + i];
    if (b < 32) sb1[b] = b1[n * 32 + b];
    if (b < 32) sw2[b] = w2[n * 32 + b];
    if (b < 2)  sb2[b] = b2[n * 2 + b];
    __syncthreads();
    float tr[32];
    const float* tp = trace + ((size_t)b * DM + n) * 32;
    #pragma unroll
    for (int m = 0; m < 32; ++m) tr[m] = tp[(m + tick + 1) & 31];   // circular -> logical
    float hbuf[16];
    #pragma unroll
    for (int j = 0; j < 16; ++j) {
        float a = sb1[j], g = sb1[j + 16];
        #pragma unroll
        for (int m = 0; m < 32; ++m) {
            a += tr[m] * sw1[m * 32 + j];
            g += tr[m] * sw1[m * 32 + j + 16];
        }
        hbuf[j] = a * (1.f / (1.f + __expf(-g)));
    }
    float o0 = sb2[0], o1 = sb2[1];
    #pragma unroll
    for (int j = 0; j < 16; ++j) { o0 += hbuf[j] * sw2[j * 2]; o1 += hbuf[j] * sw2[j * 2 + 1]; }
    act[(size_t)b * DM + n] = o0 * (1.f / (1.f + __expf(-o1)));
}

__global__ void __launch_bounds__(256)
k_sync_o_heads(const float* __restrict__ act, const int* __restrict__ i0, const int* __restrict__ i1,
               const float* __restrict__ r, const float* __restrict__ db, float* __restrict__ da,
               const float* __restrict__ Wqp, const float* __restrict__ Wcp,
               float* __restrict__ q_all, float* __restrict__ c_all, int tick)
{
    __shared__ float rq[256], rc[256];
    int s = blockIdx.x * 256 + threadIdx.x;
    int b = blockIdx.y, tid = threadIdx.x;
    float pq = 0, pc = 0;
    if (s < SRA_) {
        const float* ab = act + (size_t)b * DM;              // act[:, :NSO]
        float v = ab[i0[s]] * ab[i1[s]];
        size_t idx = (size_t)b * SRA_ + s;
        float d = r[s] * da[idx] + v;
        da[idx] = d;
        float syn = d * rsqrtf(db[s]);
        pq = syn * Wqp[s];
        pc = syn * Wcp[s];
    }
    rq[tid] = pq; rc[tid] = pc; __syncthreads();
    for (int st = 128; st > 0; st >>= 1) {
        if (tid < st) { rq[tid] += rq[tid + st]; rc[tid] += rc[tid + st]; }
        __syncthreads();
    }
    if (tid == 0) {
        atomicAdd(&q_all[b * TICKS + tick], rq[0]);
        atomicAdd(&c_all[b * TICKS + tick], rc[0]);
    }
}

__global__ void k_finalize(const float* q_all, const float* c_all,
                           const float* bqp, const float* bcp, float* out) {
    int i = blockIdx.x * 256 + threadIdx.x;
    if (i < BQ * TICKS) {
        int b = i >> 4, t = i & 15;
        out[i] = q_all[i] + bqp[0];                          // q_values (B,1,T)
        float c = c_all[i] + bcp[0];
        float cert = 1.f / (1.f + __expf(-c));               // ALPHA == 0.0
        out[BQ * TICKS + b * 32 + t]      = 1.f - cert;      // certainties (B,2,T)
        out[BQ * TICKS + b * 32 + 16 + t] = cert;
    }
}

// ---------------------------------------------------------------------------
extern "C" void kernel_launch(void* const* d_in, const int* in_sizes, int n_in,
                              void* d_out, int out_size, void* d_ws, size_t ws_size,
                              hipStream_t stream)
{
    (void)in_sizes; (void)n_in; (void)out_size; (void)ws_size;
    const float* traj   = (const float*)d_in[0];
    const float* obs    = (const float*)d_in[1];
    const float* acts   = (const float*)d_in[2];
    const float* W_traj = (const float*)d_in[3];
    const float* b_traj = (const float*)d_in[4];
    const float* W_agent= (const float*)d_in[5];
    const float* b_agent= (const float*)d_in[6];
    const float* kvg    = (const float*)d_in[7];
    const float* kvb    = (const float*)d_in[8];
    const float* W_qq   = (const float*)d_in[9];
    const float* b_qq   = (const float*)d_in[10];
    const float* Wq     = (const float*)d_in[11];
    const float* bq     = (const float*)d_in[12];
    const float* Wk     = (const float*)d_in[13];
    const float* bk     = (const float*)d_in[14];
    const float* Wv     = (const float*)d_in[15];
    const float* bv     = (const float*)d_in[16];
    const float* Wo     = (const float*)d_in[17];
    const float* bo     = (const float*)d_in[18];
    const float* W_syn  = (const float*)d_in[19];
    const float* b_syn  = (const float*)d_in[20];
    const float* syng   = (const float*)d_in[21];
    const float* synb   = (const float*)d_in[22];
    const float* sl1w   = (const float*)d_in[23];
    const float* sl1b   = (const float*)d_in[24];
    const float* sl2w   = (const float*)d_in[25];
    const float* sl2b   = (const float*)d_in[26];
    const float* W_qp   = (const float*)d_in[27];
    const float* b_qp   = (const float*)d_in[28];
    const float* W_cp   = (const float*)d_in[29];
    const float* b_cp   = (const float*)d_in[30];
    const float* sas    = (const float*)d_in[31];
    const float* st     = (const float*)d_in[32];
    const float* dpa    = (const float*)d_in[33];
    const float* dpo    = (const float*)d_in[34];
    float* out = (float*)d_out;

    // ---- workspace bump allocator ----
    size_t off = 0;
    auto alloc = [&](size_t bytes) -> char* {
        off = (off + 255) & ~(size_t)255;
        char* p = (char*)d_ws + off;
        off += bytes;
        return p;
    };
    f16*   WqqT_h  = (f16*)alloc((size_t)SRA_ * DIN * 2);    // [n][k] transposed
    f16*   WkT_h   = (f16*)alloc((size_t)DIN * DIN * 2);
    f16*   WvT_h   = (f16*)alloc((size_t)DIN * DIN * 2);
    f16*   WqT_h   = (f16*)alloc((size_t)DIN * DIN * 2);
    f16*   WoT_h   = (f16*)alloc((size_t)DIN * DIN * 2);
    f16*   WsynT_h = (f16*)alloc((size_t)DSYN * DSYN2 * 2);
    f16*   kv_h    = (f16*)alloc((size_t)BQ * SSEQ * DIN * 2);
    f16*   kh_h    = (f16*)alloc((size_t)BQ * SSEQ * DIN * 2);
    f16*   vh_h    = (f16*)alloc((size_t)BQ * SSEQ * DIN * 2);
    float* act_f   = (float*)alloc((size_t)BQ * DM * 4);
    float* trace_f = (float*)alloc((size_t)BQ * DM * 32 * 4);
    float* da_a    = (float*)alloc((size_t)BQ * SRA_ * 4);
    float* da_o    = (float*)alloc((size_t)BQ * SRA_ * 4);
    float* db_a    = (float*)alloc((size_t)SRA_ * 4);
    float* db_o    = (float*)alloc((size_t)SRA_ * 4);
    float* r_act   = (float*)alloc((size_t)SRA_ * 4);
    float* r_out   = (float*)alloc((size_t)SRA_ * 4);
    int*   i0t     = (int*)alloc((size_t)SRA_ * 4);
    int*   i1t     = (int*)alloc((size_t)SRA_ * 4);
    f16*   synch_h = (f16*)alloc((size_t)BQ * SRA_ * 2);
    float* qacc    = (float*)alloc((size_t)BQ * DIN * 4);
    f16*   qv_h    = (f16*)alloc((size_t)BQ * DIN * 2);
    float* qh_f    = (float*)alloc((size_t)BQ * DIN * 4);
    f16*   ao_h    = (f16*)alloc((size_t)BQ * DIN * 2);
    f16*   concat_h= (f16*)alloc((size_t)BQ * DSYN * 2);
    float* syn_pre = (float*)alloc((size_t)BQ * DSYN2 * 4);
    float* q_all   = (float*)alloc((size_t)BQ * TICKS * 4);
    float* c_all   = (float*)alloc((size_t)BQ * TICKS * 4);

    auto gb = [](size_t n) { return (unsigned)((n + 255) / 256); };
    auto gemm = [&](const f16* A, const f16* BT, const float* bias, float* oF, f16* oH,
                    int M, int N, int K, int ld, int ks) {
        dim3 g(N / 64, M / 64, ks);
        k_gemm_wmma<<<g, 256, 0, stream>>>(A, BT, bias, oF, oH, M, N, K, ld, ks);
    };
    auto cvtT = [&](const float* src, f16* dst, int K, int N) {
        k_cvt_f16T<<<dim3(N / 32, K / 32), dim3(32, 8), 0, stream>>>(src, dst, K, N);
    };

    // ---- setup (tick-invariant) ----
    k_tables<<<1, 256, 0, stream>>>(i0t, i1t);
    k_decay_init<<<gb(SRA_), 256, 0, stream>>>(dpa, dpo, r_act, r_out, db_a, db_o);
    k_init_state<<<gb((size_t)BQ * DM * 32), 256, 0, stream>>>(sas, st, act_f, trace_f, q_all, c_all);
    k_init_da<<<gb((size_t)BQ * SRA_), 256, 0, stream>>>(sas, i0t, i1t, da_a, da_o);
    cvtT(W_qq,  WqqT_h,  SRA_, DIN);
    cvtT(Wk,    WkT_h,   DIN,  DIN);
    cvtT(Wv,    WvT_h,   DIN,  DIN);
    cvtT(Wq,    WqT_h,   DIN,  DIN);
    cvtT(Wo,    WoT_h,   DIN,  DIN);
    cvtT(W_syn, WsynT_h, DSYN, DSYN2);
    k_build_kv<<<BQ * SSEQ, 128, 0, stream>>>(traj, obs, acts, W_traj, b_traj,
                                              W_agent, b_agent, kvg, kvb, kv_h);
    gemm(kv_h, WkT_h, bk, nullptr, kh_h, BQ * SSEQ, DIN, DIN, DIN, 1);   // K heads
    gemm(kv_h, WvT_h, bv, nullptr, vh_h, BQ * SSEQ, DIN, DIN, DIN, 1);   // V heads

    const unsigned SB = gb(SRA_);   // 129 blocks over the triu index space
    // ---- 16 sequential ticks ----
    for (int t = 0; t < TICKS; ++t) {
        k_db<<<SB, 256, 0, stream>>>(r_act, r_out, db_a, db_o);
        k_sync_a<<<dim3(SB, BQ), 256, 0, stream>>>(act_f, i0t, i1t, r_act, db_a, da_a, synch_h);
        k_zero_f32<<<gb((size_t)BQ * DIN), 256, 0, stream>>>(qacc, (size_t)BQ * DIN);
        gemm(synch_h, WqqT_h, nullptr, qacc, nullptr, BQ, DIN, SRA_, DIN, 8);   // split-K=8
        k_bias_cvt<<<gb((size_t)BQ * DIN), 256, 0, stream>>>(qacc, b_qq, qv_h, DIN, (size_t)BQ * DIN);
        gemm(qv_h, WqT_h, bq, qh_f, nullptr, BQ, DIN, DIN, DIN, 1);
        k_attention<<<BQ, 256, 0, stream>>>(qh_f, kh_h, vh_h, ao_h);
        gemm(ao_h, WoT_h, bo, nullptr, concat_h, BQ, DIN, DIN, DSYN, 1);        // cols [0,512)
        k_fill_act<<<gb((size_t)BQ * DM), 256, 0, stream>>>(act_f, concat_h);   // cols [512,1536)
        gemm(concat_h, WsynT_h, b_syn, syn_pre, nullptr, BQ, DSYN2, DSYN, DSYN2, 1);
        k_glu_ln_trace<<<BQ, 256, 0, stream>>>(syn_pre, syng, synb, trace_f, t & 31);
        k_trace_mlp<<<DM, 256, 0, stream>>>(trace_f, sl1w, sl1b, sl2w, sl2b, act_f, t);
        k_sync_o_heads<<<dim3(SB, BQ), 256, 0, stream>>>(act_f, i0t, i1t, r_out, db_o, da_o,
                                                         W_qp, W_cp, q_all, c_all, t);
    }
    k_finalize<<<gb((size_t)BQ * TICKS), 256, 0, stream>>>(q_all, c_all, b_qp, b_cp, out);
}
